// TNANN_30769145709146
// MI455X (gfx1250) — matrix operand order repaired and checked by reading, not compiled, not run on hardware
//
#include <hip/hip_runtime.h>
#include <hip/hip_bf16.h>
#include <math.h>

typedef __attribute__((ext_vector_type(16))) _Float16 v16h;
typedef __attribute__((ext_vector_type(8)))  _Float16 v8h;
typedef __attribute__((ext_vector_type(8)))  float    v8f;

#define VIS_N   8192
#define INP_N   4096
#define OUP_N   512
#define SEXT_N  (VIS_N + INP_N + 32)
#define WSCALE      16384.0f        // 2^14: lift W out of f16-subnormal range
#define INV_WSCALE  (1.0 / 16384.0) // exact descale of the linear GEMM result

// workspace layout (bytes)
#define OFF_SEXT  0u
#define OFF_T     (32u * 1024u)                         // float2[128*128*64]  = 8 MB
#define OFF_WH    (OFF_T + 128u*128u*64u*8u)            // f16[1024*4096]      = 8 MB
#define OFF_PART  (OFF_WH + 1024u*4096u*2u)             // double2[128*32]
#define MBLK      4                                     // M-tiles per wave (B reuse x4)
#define N_PART    (128 * 32)

// ---- s_ext[i] = 2*vis[i mod 8192] - 1 as f16 (so A[m,k] = s_ext[m+k], no modulo in GEMM)
__global__ void k_sext(const float* __restrict__ vis, _Float16* __restrict__ s_ext) {
    int i = blockIdx.x * 256 + threadIdx.x;
    if (i < SEXT_N) s_ext[i] = (_Float16)(2.0f * vis[i & (VIS_N - 1)] - 1.0f);
}

// ---- step A: T[io][jp][b] = sum_a t0[i,o,a] * t1[j,p,b,a]   (complex, f64 accumulate)
__global__ void k_stepA(const double2* __restrict__ t0, const double2* __restrict__ t1,
                        float2* __restrict__ T) {
    int tid = blockIdx.x * 256 + threadIdx.x;   // 128*128*64 threads
    int b   = tid & 63;
    int jp  = (tid >> 6) & 127;
    int io  = tid >> 13;
    const double2* p0 = t0 + io * 64;
    const double2* p1 = t1 + (jp * 64 + b) * 64;
    double ar = 0.0, ai = 0.0;
    for (int a = 0; a < 64; ++a) {
        double2 x = p0[a], y = p1[a];
        ar = fma(x.x, y.x, fma(-x.y, y.y, ar));
        ai = fma(x.x, y.y, fma( x.y, y.x, ai));
    }
    T[tid] = make_float2((float)ar, (float)ai);
}

// ---- step B: W[r,c] = sum_b T[io][jp][b] * t2[k,q,b], stored planar f16 (re rows 0..511, im rows 512..1023)
// reshape math: r = io*4 + (j>>2), c = (j&3)*1024 + p*128 + k*8 + q
__global__ void k_stepB(const float2* __restrict__ T, const double2* __restrict__ t2,
                        _Float16* __restrict__ Wh) {
    int tid = blockIdx.x * 256 + threadIdx.x;   // 512*4096 threads
    int c  = tid & 4095;
    int r  = tid >> 12;
    int io = r >> 2;
    int j  = ((r & 3) << 2) | (c >> 10);
    int p  = (c >> 7) & 7;
    int k  = (c >> 3) & 15;
    int q  = c & 7;
    const float2*  pT = T  + (io * 128 + (j * 8 + p)) * 64;
    const double2* p2 = t2 + (k * 8 + q) * 64;
    double wr = 0.0, wi = 0.0;
    for (int b = 0; b < 64; ++b) {
        float2 tv = pT[b];
        double2 u = p2[b];
        wr += (double)tv.x * u.x - (double)tv.y * u.y;
        wi += (double)tv.x * u.y + (double)tv.y * u.x;
    }
    Wh[r * 4096 + c]           = (_Float16)((float)wr * WSCALE);
    Wh[(512 + r) * 4096 + c]   = (_Float16)((float)wi * WSCALE);
}

// ---- fused GEMM + complex logcosh
// One wave per 64(M) x 16(N) super-tile, both complex planes:
// 8 WMMAs per K-step against 12 b128 loads (1.5 loads/WMMA; B L2 traffic cut 4x).
__global__ void __launch_bounds__(32)
k_gemm(const _Float16* __restrict__ s_ext, const _Float16* __restrict__ Wh,
       double2* __restrict__ partials) {
    int lane    = threadIdx.x;
    int m_super = blockIdx.x;         // 0..127  -> rows m_super*64 ..
    int n_tile  = blockIdx.y;         // 0..31   -> columns q
    int row     = lane & 15;
    int ksel    = lane >> 4;          // contiguous k order: lanes 16-31 hold K offset +16

    const _Float16* pA  = s_ext + m_super * (16 * MBLK) + row + ksel * 16;
    const _Float16* pBr = Wh + (n_tile * 16 + row) * 4096 + ksel * 16;
    const _Float16* pBi = pBr + (size_t)512 * 4096;

    v8f cre[MBLK], cim[MBLK];
    #pragma unroll
    for (int t = 0; t < MBLK; ++t) { cre[t] = (v8f){}; cim[t] = (v8f){}; }

    for (int kk = 0; kk < 128; ++kk) {
        int kb = kk * 32;
        // B fragments (32B-aligned, reused by all 4 M-tiles)
        v16h br = *(const v16h*)(pBr + kb);
        v16h bi = *(const v16h*)(pBi + kb);
        // A fragments for 4 adjacent M-tiles (2-byte-aligned +/-1 stream, WGP$-resident)
        v16h a[MBLK];
        #pragma unroll
        for (int t = 0; t < MBLK; ++t) {
            v8h alo, ahi;
            __builtin_memcpy(&alo, pA + t * 16 + kb, 16);
            __builtin_memcpy(&ahi, pA + t * 16 + kb + 8, 16);
            #pragma unroll
            for (int e = 0; e < 8; ++e) { a[t][e] = alo[e]; a[t][8 + e] = ahi[e]; }
        }
        #pragma unroll
        for (int t = 0; t < MBLK; ++t) {
            cre[t] = __builtin_amdgcn_wmma_f32_16x16x32_f16(false, a[t], false, br,
                                                            (short)0, cre[t], false, false);
            cim[t] = __builtin_amdgcn_wmma_f32_16x16x32_f16(false, a[t], false, bi,
                                                            (short)0, cim[t], false, false);
        }
    }

    // D layout: lanes 0-15 -> M=0..7 (vgpr r), lanes 16-31 -> M=8..15; N = lane&15
    double re = 0.0, im = 0.0;
    #pragma unroll
    for (int t = 0; t < MBLK; ++t) {
        int mbase = m_super * (16 * MBLK) + t * 16 + ksel * 8;
        #pragma unroll
        for (int rr = 0; rr < 8; ++rr) {
            int m = mbase + rr;
            if (m < VIS_N - 1) {                   // reference sums only 8191 shifts
                double zr = (double)cre[t][rr] * INV_WSCALE;
                double zi = (double)cim[t][rr] * INV_WSCALE;
                double wr = cosh(zr) * cos(zi);    // cosh(zr + i*zi)
                double wi = sinh(zr) * sin(zi);
                re += 0.5 * log(wr * wr + wi * wi);// Re log cosh z
                im += atan2(wi, wr);               // Im log cosh z
            }
        }
    }
    for (int off = 16; off > 0; off >>= 1) {
        re += __shfl_down(re, off, 32);
        im += __shfl_down(im, off, 32);
    }
    if (lane == 0) partials[n_tile * 128 + m_super] = make_double2(re, im);
}

// ---- deterministic final reduction of per-wave partials; writes d_out
__global__ void k_final(const double2* __restrict__ partials, float* __restrict__ out,
                        int out_size) {
    __shared__ double sre[256];
    __shared__ double sim[256];
    int tid = threadIdx.x;
    double re = 0.0, im = 0.0;
    for (int i = tid; i < N_PART; i += 256) {
        double2 v = partials[i];
        re += v.x; im += v.y;
    }
    sre[tid] = re; sim[tid] = im;
    __syncthreads();
    for (int s = 128; s > 0; s >>= 1) {
        if (tid < s) { sre[tid] += sre[tid + s]; sim[tid] += sim[tid + s]; }
        __syncthreads();
    }
    if (tid == 0) {
        out[0] = (float)sre[0];
        if (out_size > 1) out[1] = (float)sim[0];
    }
}

extern "C" void kernel_launch(void* const* d_in, const int* in_sizes, int n_in,
                              void* d_out, int out_size, void* d_ws, size_t ws_size,
                              hipStream_t stream) {
    (void)in_sizes; (void)n_in; (void)ws_size;
    const float*   vis = (const float*)d_in[0];
    const double2* t0  = (const double2*)d_in[1];   // complex128 -> interleaved double
    const double2* t1  = (const double2*)d_in[2];
    const double2* t2  = (const double2*)d_in[3];

    char* ws = (char*)d_ws;
    _Float16* s_ext = (_Float16*)(ws + OFF_SEXT);
    float2*   T     = (float2*)  (ws + OFF_T);
    _Float16* Wh    = (_Float16*)(ws + OFF_WH);
    double2*  parts = (double2*) (ws + OFF_PART);

    k_sext <<<(SEXT_N + 255) / 256, 256, 0, stream>>>(vis, s_ext);
    k_stepA<<<(128 * 128 * 64) / 256, 256, 0, stream>>>(t0, t1, T);
    k_stepB<<<(512 * 4096) / 256,    256, 0, stream>>>(T, t2, Wh);
    dim3 grid(128, 32);
    k_gemm <<<grid, 32, 0, stream>>>(s_ext, Wh, parts);
    k_final<<<1, 256, 0, stream>>>(parts, (float*)d_out, out_size);
}
